// TemporalPyramid_18957985645080
// MI455X (gfx1250) — matrix operand order repaired and checked by reading, not compiled, not run on hardware
//
#include <hip/hip_runtime.h>

// ---------------------------------------------------------------------------
// MI455X (gfx1250) temporal-pyramid windowed cross-attention.
// - All GEMMs via v_wmma_f32_16x16x32_f16 (wave32 WMMA, f32 accumulate).
// - Weights pre-transposed + converted to f16 [N][K] so every WMMA operand
//   is a contiguous 16-half per-lane load (vectorizes to global_load_b128).
// - Combine GEMM stages its 64x32 f16 weight tile into LDS with the Tensor
//   Data Mover (tensor_load_to_lds + s_wait_tensorcnt), ds_load fragments.
// Workspace usage: ~345 MB.
// ---------------------------------------------------------------------------

typedef __attribute__((ext_vector_type(16))) _Float16 v16h;
typedef __attribute__((ext_vector_type(8)))  float    v8f;
typedef __attribute__((ext_vector_type(4)))  unsigned int v4u;
typedef __attribute__((ext_vector_type(4)))  int v4i;
typedef __attribute__((ext_vector_type(8)))  int v8i;

#define Bn 2
#define Yd 128
#define Xd 128

#if __has_builtin(__builtin_amdgcn_tensor_load_to_lds) && \
    __has_builtin(__builtin_amdgcn_s_wait_tensorcnt)
#define HAVE_TDM 1
#else
#define HAVE_TDM 0
#endif

struct IdxArr { int v[9]; };

// A-operand K pattern: contiguous, K = 16*hi + e (matches B fragments)
__device__ __forceinline__ int a_kpat(int e, int hi) {
  return (hi ? 16 : 0) + e;
}
// B-operand: lane&15 = column N; lanes 0-15 K=e sequential, lanes 16-31 K=16+e.

__device__ __forceinline__ size_t tokRow(int f, int b, int wy, int wx, int tok) {
  return ((size_t)(f * Bn + b) * Yd + wy * 8 + (tok >> 3)) * Xd + wx * 8 + (tok & 7);
}

// ---------------------------------------------------------------------------
// 0) weight transpose + f32->f16: in [z][K][N] -> out [z][N][K] f16
// ---------------------------------------------------------------------------
__global__ __launch_bounds__(256) void transpose_w_kernel(
    const float* __restrict__ in, _Float16* __restrict__ out, int K, int N) {
  int z = blockIdx.y;
  const float* src = in + (size_t)z * K * N;
  _Float16* dst = out + (size_t)z * K * N;
  int idx = (int)(blockIdx.x * 256 + threadIdx.x);
  if (idx >= K * N) return;
  int k = idx / N, n = idx % N;
  dst[(size_t)n * K + k] = (_Float16)src[idx];
}

// ---------------------------------------------------------------------------
// 1) embedding + LayerNorm, f32 in -> f16 out. One wave per pixel.
// ---------------------------------------------------------------------------
template <int CIN>
__global__ __launch_bounds__(256) void prep_ln_kernel(
    const float* __restrict__ cur, const float* __restrict__ emb,
    const float* __restrict__ lng, const float* __restrict__ lnb,
    _Float16* __restrict__ outh, IdxArr eidx, int framePix, int nPix) {
  constexpr int CPL = CIN / 32;
  int gw = (int)((blockIdx.x * blockDim.x + threadIdx.x) >> 5);
  int lane = threadIdx.x & 31;
  if (gw >= nPix) return;                       // uniform per wave
  int t = gw / framePix;
  const float* p = cur + (size_t)gw * CIN;
  const float* e = emb + (size_t)eidx.v[t] * CIN;
  float vbuf[CPL];
  float s = 0.f, s2 = 0.f;
#pragma unroll
  for (int i = 0; i < CPL; ++i) {
    int c = lane + i * 32;
    float x = p[c] + e[c];
    vbuf[i] = x; s += x; s2 += x * x;
  }
#pragma unroll
  for (int m = 16; m >= 1; m >>= 1) {
    s  += __shfl_xor(s,  m, 32);
    s2 += __shfl_xor(s2, m, 32);
  }
  float mu  = s / (float)CIN;
  float var = s2 / (float)CIN - mu * mu;
  float rs  = rsqrtf(var + 1e-3f);
  _Float16* o = outh + (size_t)gw * CIN;
#pragma unroll
  for (int i = 0; i < CPL; ++i) {
    int c = lane + i * 32;
    o[c] = (_Float16)((vbuf[i] - mu) * rs * lng[c] + lnb[c]);
  }
}

// ---------------------------------------------------------------------------
// 2) per-frame Q/K/V projection: [64 x CIN] @ W^T[CIN x CIN] + bias -> f16.
//    WT is [z][n][k] f16 => contiguous B fragments.
// ---------------------------------------------------------------------------
template <int CIN>
__global__ __launch_bounds__(256) void qkv_proj_kernel(
    const _Float16* __restrict__ xin, const _Float16* __restrict__ WT,
    const float* __restrict__ bqkv, _Float16* __restrict__ qkv, int nTok) {
  constexpr int NT = CIN / 16, KT = CIN / 32, TILES = 4 * NT;
  int z = blockIdx.y;
  const _Float16* W    = WT + (size_t)z * CIN * CIN;
  const float*    bias = bqkv + (size_t)z * CIN;
  const _Float16* Arow0 = xin + (size_t)blockIdx.x * 64 * CIN;
  _Float16* Orow0 = qkv + ((size_t)z * nTok + (size_t)blockIdx.x * 64) * CIN;
  int wave = threadIdx.x >> 5, lane = threadIdx.x & 31;
  int l15 = lane & 15, hi = lane >> 4;
  for (int tid = wave; tid < TILES; tid += 8) {
    int mt = tid / NT, nt = tid % NT;
    v8f acc = {};
    const _Float16* ar = Arow0 + (size_t)(mt * 16 + l15) * CIN;
    const _Float16* br = W + (size_t)(nt * 16 + l15) * CIN + (hi ? 16 : 0);
    for (int kt = 0; kt < KT; ++kt) {
      v16h a, bf;
#pragma unroll
      for (int e2 = 0; e2 < 16; ++e2) {
        a[e2]  = ar[kt * 32 + a_kpat(e2, hi)];
        bf[e2] = br[kt * 32 + e2];
      }
      acc = __builtin_amdgcn_wmma_f32_16x16x32_f16(false, a, false, bf,
                                                   (short)0, acc, false, false);
    }
    int ncol = nt * 16 + l15;
    float bv = bias[ncol];
#pragma unroll
    for (int r = 0; r < 8; ++r) {
      int m = mt * 16 + r + hi * 8;
      Orow0[(size_t)m * CIN + ncol] = (_Float16)(acc[r] + bv);
    }
  }
}

// ---------------------------------------------------------------------------
// 3) windowed cross-attention + fused out-projection (woT = [n][k] f16).
// ---------------------------------------------------------------------------
template <int CIN>
__global__ __launch_bounds__(256) void attn_kernel(
    const _Float16* __restrict__ qkv, const _Float16* __restrict__ woT,
    const float* __restrict__ bo, _Float16* __restrict__ comb,
    IdxArr pi, IdxArr ci, IdxArr ni, int S, int nTotTok) {
  constexpr int DH  = CIN / 4;
  constexpr int NTH = DH / 16;            // AV output N tiles per head
  constexpr int KTH = (DH + 31) / 32;     // logits K tiles (zero-padded)
  __shared__ float    sL[64 * 64];
  __shared__ _Float16 sP[64 * 64];
  __shared__ _Float16 sO[64 * CIN];

  int g = blockIdx.y;
  int pair = g / (S * Bn);
  int sb = g % (S * Bn);
  int s = sb / Bn, b = sb % Bn;
  int qf, kf;
  switch (pair) {
    case 0:  qf = ci.v[s]; kf = pi.v[s]; break;   // c <- p
    case 1:  qf = pi.v[s]; kf = ci.v[s]; break;   // p <- c
    case 2:  qf = ci.v[s]; kf = ni.v[s]; break;   // c <- n
    default: qf = ni.v[s]; kf = ci.v[s]; break;   // n <- c
  }
  int wy = blockIdx.x >> 4, wx = blockIdx.x & 15;

  const _Float16* Qb = qkv;
  const _Float16* Kb = qkv + (size_t)nTotTok * CIN;
  const _Float16* Vb = Kb + (size_t)nTotTok * CIN;
  float scale = rsqrtf((float)DH);
  int wave = threadIdx.x >> 5, lane = threadIdx.x & 31;
  int l15 = lane & 15, hi = lane >> 4;

  for (int h = 0; h < 4; ++h) {
    // ---- logits = Q_h @ K_h^T * scale ----
    for (int tid = wave; tid < 16; tid += 8) {
      int mt = tid >> 2, nt = tid & 3;
      v8f acc = {};
      const _Float16* qr = Qb + tokRow(qf, b, wy, wx, mt * 16 + l15) * CIN + h * DH;
      const _Float16* kr = Kb + tokRow(kf, b, wy, wx, nt * 16 + l15) * CIN + h * DH;
      for (int kt = 0; kt < KTH; ++kt) {
        v16h a, bf;
#pragma unroll
        for (int e2 = 0; e2 < 16; ++e2) {
          int ka = kt * 32 + a_kpat(e2, hi);
          a[e2] = (ka < DH) ? qr[ka] : (_Float16)0.f;
          int kb = kt * 32 + (hi ? 16 : 0) + e2;
          bf[e2] = (kb < DH) ? kr[kb] : (_Float16)0.f;
        }
        acc = __builtin_amdgcn_wmma_f32_16x16x32_f16(false, a, false, bf,
                                                     (short)0, acc, false, false);
      }
#pragma unroll
      for (int r = 0; r < 8; ++r)
        sL[(mt * 16 + r + hi * 8) * 64 + nt * 16 + l15] = acc[r] * scale;
    }
    __syncthreads();
    // ---- softmax rows (f32 in LDS) -> f16 probs ----
    if (threadIdx.x < 64) {
      int row = threadIdx.x;
      float mx = -1e30f;
      for (int k = 0; k < 64; ++k) mx = fmaxf(mx, sL[row * 64 + k]);
      float sum = 0.f;
      for (int k = 0; k < 64; ++k) {
        float ex = __expf(sL[row * 64 + k] - mx);
        sum += ex;
        sP[row * 64 + k] = (_Float16)ex;
      }
      float inv = 1.f / sum;
      for (int k = 0; k < 64; ++k)
        sP[row * 64 + k] = (_Float16)((float)sP[row * 64 + k] * inv);
    }
    __syncthreads();
    // ---- out_h = probs @ V_h ----
    for (int tid = wave; tid < 4 * NTH; tid += 8) {
      int mt = tid / NTH, nt = tid % NTH;
      v8f acc = {};
      for (int kt = 0; kt < 2; ++kt) {
        v16h a, bf;
#pragma unroll
        for (int e2 = 0; e2 < 16; ++e2) {
          a[e2] = sP[(mt * 16 + l15) * 64 + kt * 32 + a_kpat(e2, hi)];
          int ktok = kt * 32 + (hi ? 16 : 0) + e2;
          bf[e2] = Vb[tokRow(kf, b, wy, wx, ktok) * CIN + h * DH + nt * 16 + l15];
        }
        acc = __builtin_amdgcn_wmma_f32_16x16x32_f16(false, a, false, bf,
                                                     (short)0, acc, false, false);
      }
#pragma unroll
      for (int r = 0; r < 8; ++r)
        sO[(mt * 16 + r + hi * 8) * CIN + h * DH + nt * 16 + l15] = (_Float16)acc[r];
    }
    __syncthreads();
  }

  // ---- fused out-projection: sO @ wo + bo (woT contiguous B) ----
  constexpr int NT = CIN / 16, KT = CIN / 32;
  for (int tid = wave; tid < 4 * NT; tid += 8) {
    int mt = tid / NT, nt = tid % NT;
    v8f acc = {};
    const _Float16* br = woT + (size_t)(nt * 16 + l15) * CIN + (hi ? 16 : 0);
    for (int kt = 0; kt < KT; ++kt) {
      v16h a, bf;
#pragma unroll
      for (int e2 = 0; e2 < 16; ++e2) {
        a[e2]  = sO[(mt * 16 + l15) * CIN + kt * 32 + a_kpat(e2, hi)];
        bf[e2] = br[kt * 32 + e2];
      }
      acc = __builtin_amdgcn_wmma_f32_16x16x32_f16(false, a, false, bf,
                                                   (short)0, acc, false, false);
    }
    int ncol = nt * 16 + l15;
    float bv = bo[ncol];
#pragma unroll
    for (int r = 0; r < 8; ++r) {
      int mtok = mt * 16 + r + hi * 8;
      size_t crow = ((size_t)(s * Bn + b) * Yd + wy * 8 + (mtok >> 3)) * Xd +
                    wx * 8 + (mtok & 7);
      comb[crow * (size_t)(4 * CIN) + (size_t)pair * CIN + ncol] =
          (_Float16)(acc[r] + bv);
    }
  }
}

// ---------------------------------------------------------------------------
// 4) combine: relu([rows, 4*CIN] @ cw + cb) -> f32. cwT = [COUT][4*CIN] f16.
//    Weight tile [64 x 32] staged into LDS via Tensor Data Mover per K-step.
// ---------------------------------------------------------------------------
template <int CIN, int COUT>
__global__ __launch_bounds__(256) void combine_kernel(
    const _Float16* __restrict__ comb, const _Float16* __restrict__ cwT,
    const float* __restrict__ cb, float* __restrict__ down) {
  constexpr int K = 4 * CIN, KT = K / 32;
  __shared__ _Float16 sWB[64 * 32];      // 4 KB weight tile: rows n, cols k
  int wave = threadIdx.x >> 5, lane = threadIdx.x & 31;
  int l15 = lane & 15, hi = lane >> 4;
  const _Float16* Ar = comb + (size_t)blockIdx.x * 64 * K;
  int nBase = blockIdx.y * 64;
  float* Or = down + (size_t)blockIdx.x * 64 * COUT;

  int mt0 = wave >> 2, nt0 = wave & 3;          // tile 0: waves 0..7 -> (0..1, 0..3)
  int mt1 = mt0 + 2, nt1 = nt0;                 // tile 1
  const _Float16* ap0 = Ar + (size_t)(mt0 * 16 + l15) * K;
  const _Float16* ap1 = Ar + (size_t)(mt1 * 16 + l15) * K;
  v8f acc0 = {}, acc1 = {};

  for (int kt = 0; kt < KT; ++kt) {
    // ---- stage weight tile rows [nBase, nBase+64), cols [kt*32, kt*32+32) ----
#if HAVE_TDM
    if (threadIdx.x < 32) {
      unsigned long long gaddr =
          (unsigned long long)(uintptr_t)(cwT + (size_t)nBase * K + kt * 32);
      unsigned ldsAddr = (unsigned)(uintptr_t)(&sWB[0]);
      v4u g0;
      g0[0] = 1u;                                    // count=1 (user descriptor)
      g0[1] = ldsAddr;                               // lds_addr
      g0[2] = (unsigned)(gaddr & 0xffffffffu);       // global_addr[31:0]
      g0[3] = (unsigned)((gaddr >> 32) & 0x1ffffffu) | 0x80000000u; // type=2
      unsigned td0 = (unsigned)K, td1 = (unsigned)COUT;
      unsigned long long st0 = (unsigned long long)K;
      v8i g1;
      g1[0] = (int)(1u << 16);                       // data_size = 2 bytes
      g1[1] = (int)((td0 & 0xffffu) << 16);          // tensor_dim0 lo
      g1[2] = (int)((td0 >> 16) | ((td1 & 0xffffu) << 16));
      g1[3] = (int)((td1 >> 16) | (32u << 16));      // tile_dim0 = 32
      g1[4] = (int)64u;                              // tile_dim1 = 64
      g1[5] = (int)(st0 & 0xffffffffu);              // tensor_dim0_stride
      g1[6] = (int)((st0 >> 32) & 0xffffu);
      g1[7] = 0;
      v4i g2 = {0, 0, 0, 0}, g3 = {0, 0, 0, 0};
#if defined(__clang_major__) && (__clang_major__ >= 23)
      v8i g4 = {0, 0, 0, 0, 0, 0, 0, 0};
      __builtin_amdgcn_tensor_load_to_lds(g0, g1, g2, g3, g4, 0);
#else
      __builtin_amdgcn_tensor_load_to_lds(g0, g1, g2, g3, 0);
#endif
      __builtin_amdgcn_s_wait_tensorcnt(0);
    }
    __syncthreads();
#else
    for (int i = threadIdx.x; i < 64 * 32; i += 256) {
      int n = i >> 5, k = i & 31;
      sWB[i] = cwT[(size_t)(nBase + n) * K + kt * 32 + k];
    }
    __syncthreads();
#endif
    // ---- two 16x16 output tiles per wave ----
    v16h a0, a1, b0, b1;
#pragma unroll
    for (int e2 = 0; e2 < 16; ++e2) {
      int ka = kt * 32 + a_kpat(e2, hi);
      a0[e2] = ap0[ka];
      a1[e2] = ap1[ka];
      b0[e2] = sWB[(nt0 * 16 + l15) * 32 + (hi ? 16 : 0) + e2];
      b1[e2] = sWB[(nt1 * 16 + l15) * 32 + (hi ? 16 : 0) + e2];
    }
    acc0 = __builtin_amdgcn_wmma_f32_16x16x32_f16(false, a0, false, b0,
                                                  (short)0, acc0, false, false);
    acc1 = __builtin_amdgcn_wmma_f32_16x16x32_f16(false, a1, false, b1,
                                                  (short)0, acc1, false, false);
    __syncthreads();
  }
  int nc0 = nBase + nt0 * 16 + l15;
  int nc1 = nBase + nt1 * 16 + l15;
  float bv0 = cb[nc0], bv1 = cb[nc1];
#pragma unroll
  for (int r = 0; r < 8; ++r) {
    int m0 = mt0 * 16 + r + hi * 8;
    int m1 = mt1 * 16 + r + hi * 8;
    float o0 = acc0[r] + bv0, o1 = acc1[r] + bv1;
    Or[(size_t)m0 * COUT + nc0] = o0 > 0.f ? o0 : 0.f;
    Or[(size_t)m1 * COUT + nc1] = o1 > 0.f ? o1 : 0.f;
  }
}

// ---------------------------------------------------------------------------
// Host driver
// ---------------------------------------------------------------------------
template <int CIN, int COUT>
static void run_level(const float* curIn, int Tl, int S,
                      const IdxArr& eidx, const IdxArr& pi, const IdxArr& ci,
                      const IdxArr& ni,
                      const float* emb, const float* lng, const float* lnb,
                      const float* wqkv, const float* bqkv,
                      const float* wo, const float* bo,
                      const float* cw, const float* cb,
                      _Float16* curh, _Float16* qkvb, _Float16* combb,
                      _Float16* wqkvT, _Float16* woT, _Float16* cwT,
                      float* downOut, hipStream_t stream) {
  int nPix = Tl * Bn * Yd * Xd;
  // weight transpose + f16 convert
  transpose_w_kernel<<<dim3((CIN * CIN + 255) / 256, 3), 256, 0, stream>>>(
      wqkv, wqkvT, CIN, CIN);
  transpose_w_kernel<<<dim3((CIN * CIN + 255) / 256, 1), 256, 0, stream>>>(
      wo, woT, CIN, CIN);
  transpose_w_kernel<<<dim3((4 * CIN * COUT + 255) / 256, 1), 256, 0, stream>>>(
      cw, cwT, 4 * CIN, COUT);
  prep_ln_kernel<CIN><<<(nPix + 7) / 8, 256, 0, stream>>>(
      curIn, emb, lng, lnb, curh, eidx, Bn * Yd * Xd, nPix);
  qkv_proj_kernel<CIN><<<dim3(nPix / 64, 3), 256, 0, stream>>>(
      curh, wqkvT, bqkv, qkvb, nPix);
  attn_kernel<CIN><<<dim3((Yd / 8) * (Xd / 8), 4 * S * Bn), 256, 0, stream>>>(
      qkvb, woT, bo, combb, pi, ci, ni, S, nPix);
  combine_kernel<CIN, COUT><<<dim3(S * Bn * Yd * Xd / 64, COUT / 64), 256, 0,
                              stream>>>(combb, cwT, cb, downOut);
}

extern "C" void kernel_launch(void* const* d_in, const int* in_sizes, int n_in,
                              void* d_out, int out_size, void* d_ws,
                              size_t ws_size, hipStream_t stream) {
  (void)in_sizes; (void)n_in; (void)out_size; (void)ws_size;
  const float* x = (const float*)d_in[0];
  auto P = [&](int l, int k) { return (const float*)d_in[1 + l * 9 + k]; };
  // k: 0=emb 1=lng 2=lnb 3=wqkv 4=bqkv 5=wo 6=bo 7=cw 8=cb
  float* out = (float*)d_out;

  // workspace carve-up (max over levels), 256B aligned
  char* ws = (char*)d_ws;
  size_t off = 0;
  auto carve = [&](size_t bytes) {
    void* p = ws + off;
    off += (bytes + 255) & ~(size_t)255;
    return p;
  };
  const size_t maxCurH = 20971520;         // halves: level1 163840 tok x 128
  const size_t maxComb = 50331648;         // halves: level1 98304 tok x 512
  _Float16* curh  = (_Float16*)carve(maxCurH * 2);
  _Float16* qkvb  = (_Float16*)carve(3 * maxCurH * 2);
  _Float16* combb = (_Float16*)carve(maxComb * 2);
  float*    down1 = (float*)carve((size_t)3 * Bn * Yd * Xd * 192 * 4);
  _Float16* wqkvT = (_Float16*)carve((size_t)3 * 192 * 192 * 2);
  _Float16* woT   = (_Float16*)carve((size_t)192 * 192 * 2);
  _Float16* cwT   = (_Float16*)carve((size_t)4 * 192 * 256 * 2);

  float* lvl1_out = out + (size_t)2 * Yd * Xd * 256;   // "level1" output region

  // ----- level 0: cin=64 -> cout=128, S=5, Tl=9 -----
  IdxArr e0 = {{0, 1, 2, 3, 4, 5, 6, 7, 8}};
  IdxArr p0 = {{0, 1, 3, 5, 6}}, c0 = {{1, 2, 4, 6, 7}}, n0 = {{2, 3, 5, 7, 8}};
  run_level<64, 128>(x, 9, 5, e0, p0, c0, n0,
                     P(0, 0), P(0, 1), P(0, 2), P(0, 3), P(0, 4), P(0, 5),
                     P(0, 6), P(0, 7), P(0, 8),
                     curh, qkvb, combb, wqkvT, woT, cwT, lvl1_out, stream);

  // ----- level 1: cin=128 -> cout=192, S=3, Tl=5 (reads level0 output) -----
  IdxArr e1 = {{1, 2, 4, 6, 7}};
  IdxArr p1 = {{0, 1, 2}}, c1 = {{1, 2, 3}}, n1 = {{2, 3, 4}};
  run_level<128, 192>(lvl1_out, 5, 3, e1, p1, c1, n1,
                      P(1, 0), P(1, 1), P(1, 2), P(1, 3), P(1, 4), P(1, 5),
                      P(1, 6), P(1, 7), P(1, 8),
                      curh, qkvb, combb, wqkvT, woT, cwT, down1, stream);

  // ----- level 2: cin=192 -> cout=256, S=1, Tl=3 -----
  IdxArr e2 = {{2, 4, 6}};
  IdxArr p2 = {{0}}, c2 = {{1}}, n2 = {{2}};
  run_level<192, 256>(down1, 3, 1, e2, p2, c2, n2,
                      P(2, 0), P(2, 1), P(2, 2), P(2, 3), P(2, 4), P(2, 5),
                      P(2, 6), P(2, 7), P(2, 8),
                      curh, qkvb, combb, wqkvT, woT, cwT, out, stream);
}
